// DeepseekV2MLAAttention_91147795956115
// MI455X (gfx1250) — matrix operand order repaired and checked by reading, not compiled, not run on hardware
//
#include <hip/hip_runtime.h>
#include <cstdint>
#include <cstddef>

// ---------------- problem constants (match reference) ----------------
#define T_SEQ   2048
#define HID     2048
#define NH      16
#define Q_LORA  1536
#define KV_LORA 512
#define D_NOPE  128
#define D_ROPE  64
#define D_QK    192
#define D_V     128

// ---------------- CDNA5 wave32 WMMA types ----------------
typedef __attribute__((ext_vector_type(16))) __bf16 v16bf;
typedef __attribute__((ext_vector_type(8)))  __bf16 v8bf;
typedef __attribute__((ext_vector_type(8)))  float  v8f;
typedef __attribute__((ext_vector_type(4)))  int    i32x4;

static __device__ __forceinline__ v16bf bf_combine(v8bf a, v8bf b) {
  return __builtin_shufflevector(a, b, 0,1,2,3,4,5,6,7,8,9,10,11,12,13,14,15);
}
static __device__ __forceinline__ v8f wmma_bf16(v16bf a, v16bf b, v8f c) {
  // D = A(16x32 bf16) * B(32x16 bf16) + C(16x16 f32)
  return __builtin_amdgcn_wmma_f32_16x16x32_bf16(false, a, false, b, (short)0, c, false, false);
}

// ---------------- gfx1250 async global->LDS copy (ASYNCcnt-tracked) ----------------
#ifndef __has_builtin
#define __has_builtin(x) 0
#endif
#if __has_builtin(__builtin_amdgcn_global_load_async_to_lds_b128)
#define USE_ASYNC_LDS 1
#else
#define USE_ASYNC_LDS 0
#endif

static __device__ __forceinline__ void async_copy16(const __bf16* g, __bf16* l) {
#if USE_ASYNC_LDS
  // signature (per hipcc diagnostic): (as1 int4* src, as3 int4* dst, imm offset, imm cpol)
  __builtin_amdgcn_global_load_async_to_lds_b128(
      (__attribute__((address_space(1))) i32x4*)(void*)g,
      (__attribute__((address_space(3))) i32x4*)l, 0, 0);
#else
  *(v8bf*)l = *(const v8bf*)g;   // fallback: sync copy through VGPRs
#endif
}
static __device__ __forceinline__ void async_wait_all() {
#if USE_ASYNC_LDS
#if __has_builtin(__builtin_amdgcn_s_wait_asynccnt)
  __builtin_amdgcn_s_wait_asynccnt(0);
#else
  asm volatile("s_wait_asynccnt 0x0" ::: "memory");
#endif
#endif
}

// ---------------- elementwise converters ----------------
__global__ void f32_to_bf16_kernel(const float* __restrict__ x, __bf16* __restrict__ y, size_t n) {
  size_t i = (size_t)blockIdx.x * blockDim.x + threadIdx.x;
  size_t stride = (size_t)gridDim.x * blockDim.x;
  for (; i < n; i += stride) y[i] = (__bf16)x[i];
}

// x: K x N (row-major fp32) -> y: N x K (row-major bf16)  (weight pre-transpose)
__global__ void transpose_f32_to_bf16_kernel(const float* __restrict__ x, __bf16* __restrict__ y,
                                             int K, int N) {
  size_t total = (size_t)K * N;
  size_t i = (size_t)blockIdx.x * blockDim.x + threadIdx.x;
  size_t stride = (size_t)gridDim.x * blockDim.x;
  for (; i < total; i += stride) {
    int n = (int)(i / K);
    int k = (int)(i % K);
    y[(size_t)n * K + k] = (__bf16)x[(size_t)k * N + n];
  }
}

// ---------------- RMSNorm (row-wise), fp32 in -> bf16 out ----------------
__global__ __launch_bounds__(256)
void rmsnorm_bf16_kernel(const float* __restrict__ x, int ldx,
                         const float* __restrict__ w,
                         __bf16* __restrict__ y, int W) {
  const int row = blockIdx.x;
  const float* xr = x + (size_t)row * ldx;
  float ss = 0.0f;
  for (int i = threadIdx.x; i < W; i += 256) { float v = xr[i]; ss += v * v; }
  __shared__ float red[256];
  red[threadIdx.x] = ss;
  __syncthreads();
  for (int s = 128; s > 0; s >>= 1) {
    if (threadIdx.x < s) red[threadIdx.x] += red[threadIdx.x + s];
    __syncthreads();
  }
  const float scale = rsqrtf(red[0] / (float)W + 1e-6f);
  for (int i = threadIdx.x; i < W; i += 256)
    y[(size_t)row * W + i] = (__bf16)(xr[i] * scale * w[i]);
}

// ---------------- generic bf16 WMMA GEMM, double-buffered async LDS ----------------
// C(MxN, f32, row-major) = A(MxK, bf16, row-major) * Bt^T  where Bt is NxK row-major.
// Block: 128x128 C tile, 256 threads = 8 waves in a 4(M) x 2(N) grid;
// each wave computes a 32x64 region = 2x4 WMMA subtiles; K-tile = 64 (2 WMMA steps).
#define GK        64
#define G_STRIDE  72   // 64 + 8 pad; 144B rows, 16B-aligned
#define G_TILE    (128 * G_STRIDE)
#define GEMM_LDS_BYTES (4 * G_TILE * 2)   // A0,A1,B0,B1

static __device__ __forceinline__ void gemm_stage_tiles(
    const __bf16* __restrict__ A, const __bf16* __restrict__ Bt,
    __bf16* la, __bf16* lb, int m0, int n0, int N, int K, int kbase, int tid) {
  // 128x64 bf16 per matrix = 1024 chunks of 16B; 4 chunks/thread each
#pragma unroll
  for (int it = 0; it < 4; ++it) {
    int c   = tid + it * 256;
    int r   = c >> 3;
    int col = (c & 7) << 3;
    async_copy16(&A[(size_t)(m0 + r) * K + kbase + col], &la[r * G_STRIDE + col]);
    int rr = (n0 + r < N) ? (n0 + r) : 0;  // clamp: OOB cols never stored to C
    async_copy16(&Bt[(size_t)rr * K + kbase + col], &lb[r * G_STRIDE + col]);
  }
}

__global__ __launch_bounds__(256)
void gemm_bf16_wmma_kernel(const __bf16* __restrict__ A,
                           const __bf16* __restrict__ Bt,
                           float* __restrict__ C,
                           int M, int N, int K) {
  extern __shared__ __bf16 gsm[];
  // LDS layout: [A buf0][A buf1][B buf0][B buf1]; base pointers computed per
  // iteration (no pointer arrays: LDS addrs in static initializers break lld).

  const int tid  = threadIdx.x;
  const int wid  = tid >> 5;
  const int lane = tid & 31;
  const int l    = lane & 15;
  const int hi   = lane >> 4;
  const int wm   = wid >> 1;   // 0..3
  const int wn   = wid & 1;    // 0..1
  const int m0   = blockIdx.y * 128;
  const int n0   = blockIdx.x * 128;

  v8f zero = {};
  v8f acc[2][4];
#pragma unroll
  for (int i = 0; i < 2; ++i)
#pragma unroll
    for (int j = 0; j < 4; ++j) acc[i][j] = zero;

  const int ktiles = K / GK;
  // prologue: stage tile 0 into buffer 0
  gemm_stage_tiles(A, Bt, gsm, gsm + 2 * G_TILE, m0, n0, N, K, 0, tid);
  async_wait_all();
  __syncthreads();

  for (int kt = 0; kt < ktiles; ++kt) {
    const int cur = kt & 1, nxt = cur ^ 1;
    __bf16* la_cur = gsm + cur * G_TILE;
    __bf16* lb_cur = gsm + 2 * G_TILE + cur * G_TILE;
    if (kt + 1 < ktiles) {  // fill next buffer while computing current one
      __bf16* la_nxt = gsm + nxt * G_TILE;
      __bf16* lb_nxt = gsm + 2 * G_TILE + nxt * G_TILE;
      gemm_stage_tiles(A, Bt, la_nxt, lb_nxt, m0, n0, N, K, (kt + 1) * GK, tid);
    }

#pragma unroll
    for (int kc = 0; kc < 2; ++kc) {
      v16bf afr[2], bfr[4];
#pragma unroll
      for (int i = 0; i < 2; ++i) {
        const __bf16* ar = &la_cur[(wm * 32 + i * 16 + l) * G_STRIDE + kc * 32 + hi * 8];
        afr[i] = bf_combine(*(const v8bf*)ar, *(const v8bf*)(ar + 16));
      }
#pragma unroll
      for (int j = 0; j < 4; ++j)
        bfr[j] = bf_combine(*(const v8bf*)&lb_cur[(wn * 64 + j * 16 + l) * G_STRIDE + kc * 32 + hi * 8], *(const v8bf*)&lb_cur[(wn * 64 + j * 16 + l) * G_STRIDE + kc * 32 + hi * 8 + 16]);
#pragma unroll
      for (int i = 0; i < 2; ++i)
#pragma unroll
        for (int j = 0; j < 4; ++j)
          acc[i][j] = wmma_bf16(afr[i], bfr[j], acc[i][j]);
    }
    async_wait_all();   // next tile's fill complete (this wave's share)
    __syncthreads();    // all waves: done computing cur, done filling nxt
  }

#pragma unroll
  for (int i = 0; i < 2; ++i)
#pragma unroll
    for (int j = 0; j < 4; ++j) {
      int col = n0 + wn * 64 + j * 16 + l;
      if (col < N) {
#pragma unroll
        for (int r = 0; r < 8; ++r) {
          int row = m0 + wm * 32 + i * 16 + r + hi * 8;
          C[(size_t)row * N + col] = acc[i][j][r];
        }
      }
    }
}

// ---------------- build q_full / k_full / v^T (rope + layout) ----------------
// q_full : [H][T][192] bf16   (q_nope | rope(q_pe))
// k_full : [H][T][192] bf16   (k_nope | rope(k_pe) broadcast over heads)
// v_t    : [H][128][T] bf16   (V transposed per head, Bt layout for the PV matmul)
__global__ __launch_bounds__(256)
void build_qkv_kernel(const float* __restrict__ q_raw,    // [T][H*192]
                      const float* __restrict__ kv_raw,   // [T][H*256]
                      const float* __restrict__ kva_raw,  // [T][576]
                      const int*   __restrict__ positions,
                      __bf16* __restrict__ qf,
                      __bf16* __restrict__ kf,
                      __bf16* __restrict__ vt) {
  const int t = blockIdx.x;
  __shared__ float cs[32], sn[32];
  if (threadIdx.x < 32) {
    float pos = (float)positions[t];
    float inv = __powf(10000.0f, -((float)(2 * threadIdx.x)) / 64.0f);
    float f = pos * inv;
    cs[threadIdx.x] = __cosf(f);
    sn[threadIdx.x] = __sinf(f);
  }
  __syncthreads();

  const float* qr  = q_raw  + (size_t)t * (NH * D_QK);
  const float* kvr = kv_raw + (size_t)t * (NH * (D_NOPE + D_V));
  const float* ka  = kva_raw + (size_t)t * (KV_LORA + D_ROPE);

  for (int i = threadIdx.x; i < NH * D_QK; i += 256) {
    int h = i / D_QK, d = i % D_QK;
    float qv, kv;
    if (d < D_NOPE) {
      qv = qr[h * D_QK + d];
      kv = kvr[h * (D_NOPE + D_V) + d];
    } else {
      int jj = d - D_NOPE;
      const float* xq = qr + h * D_QK + D_NOPE;
      const float* xk = ka + KV_LORA;
      if (jj < 32) {
        qv = xq[jj] * cs[jj] - xq[jj + 32] * sn[jj];
        kv = xk[jj] * cs[jj] - xk[jj + 32] * sn[jj];
      } else {
        qv = xq[jj] * cs[jj - 32] + xq[jj - 32] * sn[jj - 32];
        kv = xk[jj] * cs[jj - 32] + xk[jj - 32] * sn[jj - 32];
      }
    }
    qf[((size_t)h * T_SEQ + t) * D_QK + d] = (__bf16)qv;
    kf[((size_t)h * T_SEQ + t) * D_QK + d] = (__bf16)kv;
  }
  for (int i = threadIdx.x; i < NH * D_V; i += 256) {
    int h = i / D_V, d = i % D_V;
    vt[((size_t)h * D_V + d) * T_SEQ + t] = (__bf16)kvr[h * (D_NOPE + D_V) + D_NOPE + d];
  }
}

// ---------------- flash attention (causal, online softmax, WMMA) ----------------
// grid = (T/128, H), 256 threads = 8 waves; wave w owns 16 query rows.
// LDS: K tile 128x192 (stride 200), V^T tile 128x128 (stride 136), P staging 8x16x136.
#define K_STRIDE 200
#define V_STRIDE 136
#define FLASH_LDS_ELEMS (128 * K_STRIDE + 128 * V_STRIDE + 8 * 16 * V_STRIDE)

__global__ __launch_bounds__(256)
void flash_attn_kernel(const __bf16* __restrict__ qf,
                       const __bf16* __restrict__ kf,
                       const __bf16* __restrict__ vt,
                       __bf16* __restrict__ attn,   // [T][H*128]
                       float scale) {
  extern __shared__ __bf16 smem[];
  __bf16* k_lds = smem;
  __bf16* v_lds = k_lds + 128 * K_STRIDE;
  __bf16* p_lds = v_lds + 128 * V_STRIDE;

  const int qb = blockIdx.x, h = blockIdx.y;
  const int tid  = threadIdx.x;
  const int wid  = tid >> 5;
  const int lane = tid & 31;
  const int l    = lane & 15;
  const int hi   = lane >> 4;
  const int q0   = qb * 128;

  const __bf16* qh = qf + (size_t)h * T_SEQ * D_QK;
  const __bf16* kh = kf + (size_t)h * T_SEQ * D_QK;
  const __bf16* vh = vt + (size_t)h * D_V * T_SEQ;

  // Q fragments for this wave's 16 rows: 6 chunks of K=32 over d=192
  const int qrow = q0 + wid * 16 + l;
  v16bf qfr[6];
#pragma unroll
  for (int kc = 0; kc < 6; ++kc) {
    const __bf16* p = qh + (size_t)qrow * D_QK + kc * 32 + hi * 8;
    qfr[kc] = bf_combine(*(const v8bf*)p, *(const v8bf*)(p + 16));
  }

  v8f zero = {};
  v8f o[8];
#pragma unroll
  for (int c = 0; c < 8; ++c) o[c] = zero;
  float rmax[8], rsum[8];
#pragma unroll
  for (int r = 0; r < 8; ++r) { rmax[r] = -3.0e38f; rsum[r] = 0.0f; }

  const int nkv = qb + 1;  // causal: kv blocks 0..qb
  for (int j = 0; j < nkv; ++j) {
    const int kv0 = j * 128;
    __syncthreads();
    // async-stage K tile: 128 x 192 bf16 (3072 16B chunks / 256 thr = 12 iters)
    for (int c = tid; c < 128 * 24; c += 256) {
      int r = c / 24, cc = (c % 24) * 8;
      async_copy16(&kh[(size_t)(kv0 + r) * D_QK + cc], &k_lds[r * K_STRIDE + cc]);
    }
    // async-stage V^T tile: 128(d) x 128(kv)
    for (int c = tid; c < 128 * 16; c += 256) {
      int r = c / 16, cc = (c % 16) * 8;
      async_copy16(&vh[(size_t)r * T_SEQ + kv0 + cc], &v_lds[r * V_STRIDE + cc]);
    }
    if (j + 1 < nkv)  // touch next K tile into L2 (global_prefetch_b8)
      __builtin_prefetch(&kh[(size_t)(kv0 + 128) * D_QK], 0, 1);
    async_wait_all();
    __syncthreads();

    // ---- S = Q K^T for this wave's 16 rows x 128 kv cols ----
    v8f s[8];
#pragma unroll
    for (int n = 0; n < 8; ++n) s[n] = zero;
#pragma unroll
    for (int kc = 0; kc < 6; ++kc) {
#pragma unroll
      for (int n = 0; n < 8; ++n) {
        v16bf b = bf_combine(*(const v8bf*)&k_lds[(n * 16 + l) * K_STRIDE + kc * 32 + hi * 8], *(const v8bf*)&k_lds[(n * 16 + l) * K_STRIDE + kc * 32 + hi * 8 + 16]);
        s[n] = wmma_bf16(qfr[kc], b, s[n]);
      }
    }

    // ---- scale + causal mask ----
    const bool diag = (j == qb);
#pragma unroll
    for (int n = 0; n < 8; ++n) {
#pragma unroll
      for (int r = 0; r < 8; ++r) {
        float v = s[n][r] * scale;
        if (diag) {
          int col = kv0 + n * 16 + l;
          int row = q0 + wid * 16 + r + 8 * hi;
          if (col > row) v = -3.0e38f;
        }
        s[n][r] = v;
      }
    }

    // ---- online softmax (row stats across 16 lanes of this half-wave) ----
#pragma unroll
    for (int r = 0; r < 8; ++r) {
      float tmax = -3.0e38f;
#pragma unroll
      for (int n = 0; n < 8; ++n) tmax = fmaxf(tmax, s[n][r]);
      tmax = fmaxf(tmax, __shfl_xor(tmax, 8, 16));
      tmax = fmaxf(tmax, __shfl_xor(tmax, 4, 16));
      tmax = fmaxf(tmax, __shfl_xor(tmax, 2, 16));
      tmax = fmaxf(tmax, __shfl_xor(tmax, 1, 16));
      float mnew = fmaxf(rmax[r], tmax);
      float alpha = __expf(rmax[r] - mnew);
      rmax[r] = mnew;
      float psum = 0.0f;
#pragma unroll
      for (int n = 0; n < 8; ++n) {
        float p = __expf(s[n][r] - mnew);
        s[n][r] = p;
        psum += p;
      }
      psum += __shfl_xor(psum, 8, 16);
      psum += __shfl_xor(psum, 4, 16);
      psum += __shfl_xor(psum, 2, 16);
      psum += __shfl_xor(psum, 1, 16);
      rsum[r] = rsum[r] * alpha + psum;
#pragma unroll
      for (int c = 0; c < 8; ++c) o[c][r] *= alpha;
    }

    // ---- P (C-layout regs) -> LDS -> A-layout frags; O += P V ----
    __bf16* pw = p_lds + wid * 16 * V_STRIDE;
#pragma unroll
    for (int n = 0; n < 8; ++n)
#pragma unroll
      for (int r = 0; r < 8; ++r)
        pw[(r + 8 * hi) * V_STRIDE + n * 16 + l] = (__bf16)s[n][r];

#pragma unroll
    for (int kc2 = 0; kc2 < 4; ++kc2) {
      const __bf16* pr = pw + l * V_STRIDE + kc2 * 32 + hi * 8;
      v16bf pa = bf_combine(*(const v8bf*)pr, *(const v8bf*)(pr + 16));
#pragma unroll
      for (int c = 0; c < 8; ++c) {
        v16bf vb = bf_combine(*(const v8bf*)&v_lds[(c * 16 + l) * V_STRIDE + kc2 * 32 + hi * 8], *(const v8bf*)&v_lds[(c * 16 + l) * V_STRIDE + kc2 * 32 + hi * 8 + 16]);
        o[c] = wmma_bf16(pa, vb, o[c]);
      }
    }
  }

  // ---- normalize and emit attn rows (bf16, head-major layout [T][H*128]) ----
  float inv[8];
#pragma unroll
  for (int r = 0; r < 8; ++r) inv[r] = 1.0f / rsum[r];
#pragma unroll
  for (int c = 0; c < 8; ++c)
#pragma unroll
    for (int r = 0; r < 8; ++r) {
      int row = q0 + wid * 16 + r + 8 * hi;
      int col = h * D_V + c * 16 + l;
      attn[(size_t)row * (NH * D_V) + col] = (__bf16)(o[c][r] * inv[r]);
    }
}

// ---------------- host-side orchestration ----------------
static inline size_t align_up(size_t x) { return (x + 255) & ~(size_t)255; }
static inline int cdiv(int a, int b) { return (a + b - 1) / b; }

extern "C" void kernel_launch(void* const* d_in, const int* in_sizes, int n_in,
                              void* d_out, int out_size, void* d_ws, size_t ws_size,
                              hipStream_t stream) {
  (void)in_sizes; (void)n_in; (void)out_size; (void)ws_size;
  const float* hidden    = (const float*)d_in[0];
  const int*   positions = (const int*)d_in[1];
  const float* w_qa      = (const float*)d_in[2];
  const float* q_a_ln_w  = (const float*)d_in[3];
  const float* w_qb      = (const float*)d_in[4];
  const float* w_kva     = (const float*)d_in[5];
  const float* kv_a_ln_w = (const float*)d_in[6];
  const float* w_kvb     = (const float*)d_in[7];
  const float* w_o       = (const float*)d_in[8];
  float* out = (float*)d_out;

  // workspace carve-out
  char* base = (char*)d_ws;
  size_t off = 0;
  auto carve = [&](size_t bytes) { char* p = base + off; off += align_up(bytes); return p; };

  __bf16* hbf     = (__bf16*)carve((size_t)T_SEQ * HID * 2);
  __bf16* wqaT    = (__bf16*)carve((size_t)Q_LORA * HID * 2);
  __bf16* wqbT    = (__bf16*)carve((size_t)(NH * D_QK) * Q_LORA * 2);
  __bf16* wkvaT   = (__bf16*)carve((size_t)(KV_LORA + D_ROPE) * HID * 2);
  __bf16* wkvbT   = (__bf16*)carve((size_t)(NH * (D_NOPE + D_V)) * KV_LORA * 2);
  __bf16* woT     = (__bf16*)carve((size_t)HID * (NH * D_V) * 2);
  float*  q_c_raw = (float*)carve((size_t)T_SEQ * Q_LORA * 4);
  __bf16* q_c_bf  = (__bf16*)carve((size_t)T_SEQ * Q_LORA * 2);
  float*  q_raw   = (float*)carve((size_t)T_SEQ * NH * D_QK * 4);
  float*  kva_raw = (float*)carve((size_t)T_SEQ * (KV_LORA + D_ROPE) * 4);
  __bf16* kv_c_bf = (__bf16*)carve((size_t)T_SEQ * KV_LORA * 2);
  float*  kv_raw  = (float*)carve((size_t)T_SEQ * NH * (D_NOPE + D_V) * 4);
  __bf16* q_full  = (__bf16*)carve((size_t)NH * T_SEQ * D_QK * 2);
  __bf16* k_full  = (__bf16*)carve((size_t)NH * T_SEQ * D_QK * 2);
  __bf16* v_t     = (__bf16*)carve((size_t)NH * D_V * T_SEQ * 2);
  __bf16* attn_bf = (__bf16*)carve((size_t)T_SEQ * NH * D_V * 2);

  // 1) precision conversion + weight pre-transpose (to N x K bf16 for Bt layout)
  f32_to_bf16_kernel<<<2048, 256, 0, stream>>>(hidden, hbf, (size_t)T_SEQ * HID);
  transpose_f32_to_bf16_kernel<<<cdiv(HID * Q_LORA, 256), 256, 0, stream>>>(w_qa, wqaT, HID, Q_LORA);
  transpose_f32_to_bf16_kernel<<<cdiv(Q_LORA * NH * D_QK, 256), 256, 0, stream>>>(w_qb, wqbT, Q_LORA, NH * D_QK);
  transpose_f32_to_bf16_kernel<<<cdiv(HID * (KV_LORA + D_ROPE), 256), 256, 0, stream>>>(w_kva, wkvaT, HID, KV_LORA + D_ROPE);
  transpose_f32_to_bf16_kernel<<<cdiv(KV_LORA * NH * (D_NOPE + D_V), 256), 256, 0, stream>>>(w_kvb, wkvbT, KV_LORA, NH * (D_NOPE + D_V));
  transpose_f32_to_bf16_kernel<<<cdiv(NH * D_V * HID, 256), 256, 0, stream>>>(w_o, woT, NH * D_V, HID);

  // 2) q_a projection + RMSNorm
  gemm_bf16_wmma_kernel<<<dim3(cdiv(Q_LORA, 128), T_SEQ / 128), 256, GEMM_LDS_BYTES, stream>>>(
      hbf, wqaT, q_c_raw, T_SEQ, Q_LORA, HID);
  rmsnorm_bf16_kernel<<<T_SEQ, 256, 0, stream>>>(q_c_raw, Q_LORA, q_a_ln_w, q_c_bf, Q_LORA);

  // 3) q_b projection
  gemm_bf16_wmma_kernel<<<dim3(cdiv(NH * D_QK, 128), T_SEQ / 128), 256, GEMM_LDS_BYTES, stream>>>(
      q_c_bf, wqbT, q_raw, T_SEQ, NH * D_QK, Q_LORA);

  // 4) kv_a projection (N=576, tail-clamped) + RMSNorm of first 512
  gemm_bf16_wmma_kernel<<<dim3(cdiv(KV_LORA + D_ROPE, 128), T_SEQ / 128), 256, GEMM_LDS_BYTES, stream>>>(
      hbf, wkvaT, kva_raw, T_SEQ, KV_LORA + D_ROPE, HID);
  rmsnorm_bf16_kernel<<<T_SEQ, 256, 0, stream>>>(kva_raw, KV_LORA + D_ROPE, kv_a_ln_w, kv_c_bf, KV_LORA);

  // 5) kv_b projection
  gemm_bf16_wmma_kernel<<<dim3(cdiv(NH * (D_NOPE + D_V), 128), T_SEQ / 128), 256, GEMM_LDS_BYTES, stream>>>(
      kv_c_bf, wkvbT, kv_raw, T_SEQ, NH * (D_NOPE + D_V), KV_LORA);

  // 6) RoPE + per-head Q/K assembly + V transpose
  build_qkv_kernel<<<T_SEQ, 256, 0, stream>>>(q_raw, kv_raw, kva_raw, positions,
                                              q_full, k_full, v_t);

  // 7) causal flash attention per (query-block, head)
  const float scale = 0.0721687836487032f;  // 192^-0.5
  flash_attn_kernel<<<dim3(T_SEQ / 128, NH), 256, FLASH_LDS_ELEMS * sizeof(__bf16), stream>>>(
      q_full, k_full, v_t, attn_bf, scale);

  // 8) output projection -> d_out (fp32)
  gemm_bf16_wmma_kernel<<<dim3(cdiv(HID, 128), T_SEQ / 128), 256, GEMM_LDS_BYTES, stream>>>(
      attn_bf, woT, out, T_SEQ, HID, NH * D_V);
}
